// TriangleAttention_7232724927144
// MI455X (gfx1250) — matrix operand order repaired and checked by reading, not compiled, not run on hardware
//
#include <hip/hip_runtime.h>

#define NRES 384
#define CZ   128
#define NHEAD 4
#define DHEAD 32
#define NN (NRES * NRES)

// Padded LDS strides (in halfs): DWORD stride = 4*k with k odd => 16 lanes
// reading 16 consecutive rows hit banks 4*(k*i mod 16): conflict-free.
#define XSTR 136   // 68 DW = 4*17
#define KSTR 136
#define VSTR 392   // 196 DW = 4*49
#define QSTR 32    // cold (once per task)
#define PSTR 40    // 20 DW = 4*5, hot
#define SCRW (16 * QSTR + 16 * PSTR)   // 1152 halfs per wave
#define ASTR 136   // out_kernel A block stride

typedef _Float16 h16v __attribute__((ext_vector_type(16)));
typedef _Float16 h8v  __attribute__((ext_vector_type(8)));
typedef _Float16 h4v  __attribute__((ext_vector_type(4)));
typedef float    f8v  __attribute__((ext_vector_type(8)));
typedef unsigned int u32x4 __attribute__((ext_vector_type(4)));
typedef unsigned int u32x8 __attribute__((ext_vector_type(8)));

__device__ __forceinline__ h16v cat16(h8v lo, h8v hi) {
  return __builtin_shufflevector(lo, hi, 0,1,2,3,4,5,6,7,8,9,10,11,12,13,14,15);
}

// A-operand (16x32 f16): lane L holds row L&15; K runs at offsets hs*8 and
// 16+hs*8 where hs = L>>4 (per ISA 16-bit A layout table).
__device__ __forceinline__ h16v ld_a(const _Float16* base, int row0, int stride,
                                     int k0, int lane) {
  const _Float16* p = base + (size_t)(row0 + (lane & 15)) * stride + k0 + ((lane >> 4) << 3);
  h8v lo = *(const h8v*)p;
  h8v hi = *(const h8v*)(p + 16);
  return cat16(lo, hi);
}

// B-operand (32x16 f16): lane L holds column L&15; K runs at offsets hs*8 and
// 16+hs*8 where hs = L>>4 (same K order as ld_a).
__device__ __forceinline__ h16v ld_b(const _Float16* base, int col0, int stride,
                                     int k0, int lane) {
  const _Float16* p = base + (size_t)(col0 + (lane & 15)) * stride + k0 + ((lane >> 4) << 3);
  h8v lo = *(const h8v*)p;
  h8v hi = *(const h8v*)(p + 16);
  return cat16(lo, hi);
}

__device__ __forceinline__ f8v wmma32(h16v a, h16v b, f8v c) {
  return __builtin_amdgcn_wmma_f32_16x16x32_f16(false, a, false, b, (short)0, c, false, false);
}

__device__ __forceinline__ void lds_fence() {
  asm volatile("s_wait_dscnt 0x0" ::: "memory");
}

__device__ __forceinline__ float rmax16(float v) {
  v = fmaxf(v, __shfl_xor(v, 1, 32));
  v = fmaxf(v, __shfl_xor(v, 2, 32));
  v = fmaxf(v, __shfl_xor(v, 4, 32));
  v = fmaxf(v, __shfl_xor(v, 8, 32));
  return v;
}
__device__ __forceinline__ float rsum16(float v) {
  v += __shfl_xor(v, 1, 32);
  v += __shfl_xor(v, 2, 32);
  v += __shfl_xor(v, 4, 32);
  v += __shfl_xor(v, 8, 32);
  return v;
}
__device__ __forceinline__ float rsum32(float v) {
  v += __shfl_xor(v, 1, 32);
  v += __shfl_xor(v, 2, 32);
  v += __shfl_xor(v, 4, 32);
  v += __shfl_xor(v, 8, 32);
  v += __shfl_xor(v, 16, 32);
  return v;
}

// One TDM DMA of nelem2b 2-byte elements, contiguous in memory, into LDS at
// lds_off. With pad=true, inserts 4 DWORDs (8 halfs) of LDS padding every
// 64 DWORDs (128 halfs): a [R][128] tile lands with row stride 136 halfs.
__device__ __forceinline__ void tdm_load_1d(unsigned lds_off, const void* gaddr,
                                            unsigned nelem2b, bool pad) {
  unsigned long long ga = (unsigned long long)gaddr;
  u32x4 g0;
  g0[0] = 1u;                                      // count=1, user descriptor
  g0[1] = lds_off;                                 // lds_addr (bytes)
  g0[2] = (unsigned)ga;                            // global_addr[31:0]
  g0[3] = (unsigned)((ga >> 32) & 0x1FFFFFFu) | (2u << 30);  // addr[56:32], type=2
  u32x8 g1;
  // data_size=1 (2B); pad_enable(b20), pad_interval=5 (b24:22), pad_amount=3 (b31:25)
  g1[0] = 0x00010000u | (pad ? ((1u << 20) | (5u << 22) | (3u << 25)) : 0u);
  g1[1] = (nelem2b & 0xFFFFu) << 16;               // tensor_dim0[15:0]
  g1[2] = ((nelem2b >> 16) & 0xFFFFu) | (1u << 16); // tensor_dim0[31:16], tensor_dim1=1
  g1[3] = (nelem2b & 0xFFFFu) << 16;               // tile_dim0
  g1[4] = 1u;                                      // tile_dim1=1
  g1[5] = nelem2b;                                 // tensor_dim0_stride[31:0]
  g1[6] = 0u;
  g1[7] = 0u;
  asm volatile("tensor_load_to_lds %0, %1" :: "s"(g0), "s"(g1) : "memory");
}

// ---------------------------------------------------------------------------
// Kernel 1: LayerNorm over last dim (128), write f16 x to workspace.
// ---------------------------------------------------------------------------
__global__ __launch_bounds__(256) void ln_kernel(const float* __restrict__ pa,
                                                 const float* __restrict__ g,
                                                 const float* __restrict__ b,
                                                 _Float16* __restrict__ Xh) {
  int row = blockIdx.x * 8 + (threadIdx.x >> 5);
  int lane = threadIdx.x & 31;
  const float4 v = *(const float4*)(pa + (size_t)row * CZ + lane * 4);
  float s = v.x + v.y + v.z + v.w;
  float s2 = v.x * v.x + v.y * v.y + v.z * v.z + v.w * v.w;
  s = rsum32(s);
  s2 = rsum32(s2);
  float mu = s * (1.0f / CZ);
  float var = s2 * (1.0f / CZ) - mu * mu;
  float rs = rsqrtf(var + 1e-5f);
  int c = lane * 4;
  h4v o;
  o[0] = (_Float16)((v.x - mu) * rs * g[c + 0] + b[c + 0]);
  o[1] = (_Float16)((v.y - mu) * rs * g[c + 1] + b[c + 1]);
  o[2] = (_Float16)((v.z - mu) * rs * g[c + 2] + b[c + 2]);
  o[3] = (_Float16)((v.w - mu) * rs * g[c + 3] + b[c + 3]);
  *(h4v*)(Xh + (size_t)row * CZ + c) = o;
}

// ---------------------------------------------------------------------------
// Kernel 2: weight prep — transpose 128x128 f32 -> f16 [out][in].
// ---------------------------------------------------------------------------
__global__ __launch_bounds__(256) void wprep_kernel(
    const float* __restrict__ qw, const float* __restrict__ kw,
    const float* __restrict__ vw, const float* __restrict__ gw,
    const float* __restrict__ ow, _Float16* __restrict__ dst) {
  int idx = blockIdx.x * 256 + threadIdx.x;   // 0 .. 5*16384-1
  int mat = idx >> 14;
  int e = idx & 16383;
  int o = e >> 7;
  int c = e & 127;
  const float* src;
  switch (mat) {
    case 0: src = qw; break;
    case 1: src = kw; break;
    case 2: src = vw; break;
    case 3: src = gw; break;
    default: src = ow; break;
  }
  dst[(size_t)mat * 16384 + e] = (_Float16)src[c * 128 + o];
}

// ---------------------------------------------------------------------------
// Kernel 3: nonbatched bias  NB[h][q][k] = sum_c x[q,k,c] * w2d[c,h]
// ---------------------------------------------------------------------------
__global__ __launch_bounds__(256) void bias_kernel(const _Float16* __restrict__ Xh,
                                                   const float* __restrict__ w2d,
                                                   float* __restrict__ NB) {
  int idx = blockIdx.x * 256 + threadIdx.x;  // q*NRES + k
  const h8v* xp8 = (const h8v*)(Xh + (size_t)idx * CZ);
  float a0 = 0.f, a1 = 0.f, a2 = 0.f, a3 = 0.f;
#pragma unroll 4
  for (int cc = 0; cc < CZ / 8; ++cc) {
    h8v xv = xp8[cc];
#pragma unroll
    for (int j = 0; j < 8; ++j) {
      float x = (float)xv[j];
      const float* w = w2d + (cc * 8 + j) * NHEAD;
      a0 += x * w[0];
      a1 += x * w[1];
      a2 += x * w[2];
      a3 += x * w[3];
    }
  }
  NB[0 * (size_t)NN + idx] = a0;
  NB[1 * (size_t)NN + idx] = a1;
  NB[2 * (size_t)NN + idx] = a2;
  NB[3 * (size_t)NN + idx] = a3;
}

// ---------------------------------------------------------------------------
// Kernel 4: fused gated attention per row b.
//   LDS: XS [384][XSTR], KS [384][KSTR], VTS [128][VSTR], per-wave scratch.
//   Total = (384*136 + 384*136 + 128*392 + 8*1152) halfs = 320 KB exactly.
// ---------------------------------------------------------------------------
__global__ __launch_bounds__(256) void attn_kernel(
    const _Float16* __restrict__ Xh, const float* __restrict__ mask,
    const float* __restrict__ NB, const _Float16* __restrict__ wq,
    const _Float16* __restrict__ wk, const _Float16* __restrict__ wv,
    const _Float16* __restrict__ wg, const float* __restrict__ gb,
    _Float16* __restrict__ WA) {
  extern __shared__ _Float16 smem[];
  _Float16* XS = smem;                        // [384][XSTR]
  _Float16* KS = XS + NRES * XSTR;            // [384][KSTR]
  _Float16* VTS = KS + NRES * KSTR;           // [128][VSTR]
  _Float16* SCR = VTS + CZ * VSTR;            // 8 waves * SCRW halfs

  const int b = blockIdx.x;
  const int tid = threadIdx.x;
  const int wave = tid >> 5;
  const int lane = tid & 31;
  const int ln16 = lane & 15;
  const int rbase = (lane >> 4) << 3;

  // Stage x_b into LDS with one padded TDM DMA (96 KB -> 136-half stride).
  if (wave == 0) {
    tdm_load_1d(/*lds_off=*/0, Xh + (size_t)b * NRES * CZ,
                /*nelem2b=*/NRES * CZ, /*pad=*/true);
    __builtin_amdgcn_s_wait_tensorcnt(0);
  }
  __syncthreads();

  // ---- Phase 2: K = x@k_w (KS) and V = x@v_w (VTS, transposed) fused:
  //      shared A loads, two independent WMMA chains per task.
  for (int t = wave; t < 192; t += 8) {
    const int mt = t >> 3;   // 0..23 row tile
    const int nt = t & 7;    // 0..7 col tile
    f8v ak = {}, av = {};
#pragma unroll
    for (int kc = 0; kc < 4; ++kc) {
      h16v A = ld_a(XS, mt * 16, XSTR, kc * 32, lane);
      h16v Bk = ld_b(wk, nt * 16, CZ, kc * 32, lane);
      h16v Bv = ld_b(wv, nt * 16, CZ, kc * 32, lane);
      ak = wmma32(A, Bk, ak);
      av = wmma32(A, Bv, av);
    }
    const int col = nt * 16 + ln16;
#pragma unroll
    for (int r = 0; r < 8; ++r) {
      const int row = mt * 16 + rbase + r;
      KS[(size_t)row * KSTR + col] = (_Float16)ak[r];
      VTS[(size_t)col * VSTR + row] = (_Float16)av[r];
    }
  }
  __syncthreads();

  // ---- Phase 3: flash attention per (q-tile, head) ----
  _Float16* QS = SCR + wave * SCRW;          // [16][QSTR]
  _Float16* PS = QS + 16 * QSTR;             // [16][PSTR]
  const float scale = 0.17677669529663687f;  // 1/sqrt(32)

  for (int t = wave; t < 96; t += 8) {
    const int qt = t >> 2;
    const int h = t & 3;
    const int hc = h * DHEAD;

    // Q tile (16x32) and gate tile via WMMA over c (4 independent chains).
    f8v cq0 = {}, cq1 = {}, cg0 = {}, cg1 = {};
#pragma unroll
    for (int kc = 0; kc < 4; ++kc) {
      h16v A = ld_a(XS, qt * 16, XSTR, kc * 32, lane);
      h16v Bq0 = ld_b(wq, hc, CZ, kc * 32, lane);
      h16v Bq1 = ld_b(wq, hc + 16, CZ, kc * 32, lane);
      cq0 = wmma32(A, Bq0, cq0);
      cq1 = wmma32(A, Bq1, cq1);
      h16v Bg0 = ld_b(wg, hc, CZ, kc * 32, lane);
      h16v Bg1 = ld_b(wg, hc + 16, CZ, kc * 32, lane);
      cg0 = wmma32(A, Bg0, cg0);
      cg1 = wmma32(A, Bg1, cg1);
    }
    const float gb0 = gb[hc + ln16];
    const float gb1 = gb[hc + 16 + ln16];
    f8v g0, g1;
#pragma unroll
    for (int r = 0; r < 8; ++r) {
      g0[r] = __builtin_amdgcn_rcpf(1.0f + __expf(-(cg0[r] + gb0)));
      g1[r] = __builtin_amdgcn_rcpf(1.0f + __expf(-(cg1[r] + gb1)));
    }
    // C-layout -> A-layout via LDS scratch (scaled f16 Q).
#pragma unroll
    for (int r = 0; r < 8; ++r) {
      const int row = rbase + r;
      QS[row * QSTR + ln16] = (_Float16)(cq0[r] * scale);
      QS[row * QSTR + 16 + ln16] = (_Float16)(cq1[r] * scale);
    }
    lds_fence();
    h16v Qa = ld_a(QS, 0, QSTR, 0, lane);

    f8v o0 = {}, o1 = {}, mrow, lrow = {};
#pragma unroll
    for (int r = 0; r < 8; ++r) mrow[r] = -3.0e38f;

    const int q0 = qt * 16;
    for (int kt = 0; kt < NRES / 32; ++kt) {
      const int k0 = kt * 32;
      // Bias tile (mask + nonbatched) loaded directly into WMMA C operand:
      // C/D layout row = q (VGPR r), col = k (lane).
      const float mb0 = 1e9f * (mask[(size_t)b * NRES + k0 + ln16] - 1.0f);
      const float mb1 = 1e9f * (mask[(size_t)b * NRES + k0 + 16 + ln16] - 1.0f);
      f8v c0, c1;
#pragma unroll
      for (int r = 0; r < 8; ++r) {
        const float* nbp = NB + (size_t)h * NN + (size_t)(q0 + rbase + r) * NRES;
        c0[r] = mb0 + nbp[k0 + ln16];
        c1[r] = mb1 + nbp[k0 + 16 + ln16];
      }
      // S = Q Kt + bias (two 16x16 tiles, bias as accumulator input)
      h16v Bk0 = ld_b(KS, k0, KSTR, hc, lane);
      h16v Bk1 = ld_b(KS, k0 + 16, KSTR, hc, lane);
      f8v s0 = wmma32(Qa, Bk0, c0);
      f8v s1 = wmma32(Qa, Bk1, c1);
      // online softmax update
#pragma unroll
      for (int r = 0; r < 8; ++r) {
        float tm = rmax16(fmaxf(s0[r], s1[r]));
        float nm = fmaxf(mrow[r], tm);
        float al = __expf(mrow[r] - nm);
        s0[r] = __expf(s0[r] - nm);
        s1[r] = __expf(s1[r] - nm);
        float rs = rsum16(s0[r] + s1[r]);
        lrow[r] = lrow[r] * al + rs;
        o0[r] *= al;
        o1[r] *= al;
        mrow[r] = nm;
      }
      // P f32 C-layout -> f16 A-layout via scratch
#pragma unroll
      for (int r = 0; r < 8; ++r) {
        const int row = rbase + r;
        PS[row * PSTR + ln16] = (_Float16)s0[r];
        PS[row * PSTR + 16 + ln16] = (_Float16)s1[r];
      }
      lds_fence();
      h16v Pa = ld_a(PS, 0, PSTR, 0, lane);
      h16v Bv0 = ld_b(VTS, hc, VSTR, k0, lane);
      h16v Bv1 = ld_b(VTS, hc + 16, VSTR, k0, lane);
      o0 = wmma32(Pa, Bv0, o0);
      o1 = wmma32(Pa, Bv1, o1);
    }
    // normalize, gate, store gated wa (f16) to workspace
#pragma unroll
    for (int r = 0; r < 8; ++r) {
      const float inv = __builtin_amdgcn_rcpf(lrow[r]);
      const size_t qrow = (size_t)b * NRES + q0 + rbase + r;
      WA[qrow * CZ + hc + ln16] = (_Float16)(o0[r] * inv * g0[r]);
      WA[qrow * CZ + hc + 16 + ln16] = (_Float16)(o1[r] * inv * g1[r]);
    }
  }
}

// ---------------------------------------------------------------------------
// Kernel 5: output projection  out = WA @ o_w + o_b   (147456x128 @ 128x128)
// A block staged by padded TDM DMA (32 KB -> 136-half stride, conflict-free).
// ---------------------------------------------------------------------------
__global__ __launch_bounds__(256) void out_kernel(const _Float16* __restrict__ WA,
                                                  const _Float16* __restrict__ wo,
                                                  const float* __restrict__ ob,
                                                  float* __restrict__ out) {
  __shared__ _Float16 AS[128 * ASTR];
  const int tid = threadIdx.x;
  const int wave = tid >> 5;
  const int lane = tid & 31;
  const int ln16 = lane & 15;
  const int rbase = (lane >> 4) << 3;
  const int row00 = blockIdx.x * 128;
  if (wave == 0) {
    tdm_load_1d(/*lds_off=*/0, WA + (size_t)row00 * CZ,
                /*nelem2b=*/128 * CZ, /*pad=*/true);
    __builtin_amdgcn_s_wait_tensorcnt(0);
  }
  __syncthreads();
  for (int t = wave; t < 64; t += 8) {
    const int mt = t >> 3;
    const int nt = t & 7;
    f8v acc = {};
#pragma unroll
    for (int kc = 0; kc < 4; ++kc) {
      h16v A = ld_a(AS, mt * 16, ASTR, kc * 32, lane);
      h16v B = ld_b(wo, nt * 16, CZ, kc * 32, lane);
      acc = wmma32(A, B, acc);
    }
    const int col = nt * 16 + ln16;
    const float obv = ob[col];
#pragma unroll
    for (int r = 0; r < 8; ++r) {
      out[(size_t)(row00 + mt * 16 + rbase + r) * CZ + col] = acc[r] + obv;
    }
  }
}

// ---------------------------------------------------------------------------
extern "C" void kernel_launch(void* const* d_in, const int* in_sizes, int n_in,
                              void* d_out, int out_size, void* d_ws, size_t ws_size,
                              hipStream_t stream) {
  const float* pair_act  = (const float*)d_in[0];
  const float* pair_mask = (const float*)d_in[1];
  const float* ln_g      = (const float*)d_in[2];
  const float* ln_b      = (const float*)d_in[3];
  const float* w2d       = (const float*)d_in[4];
  const float* q_w       = (const float*)d_in[5];
  const float* k_w       = (const float*)d_in[6];
  const float* v_w       = (const float*)d_in[7];
  const float* g_w       = (const float*)d_in[8];
  const float* g_b       = (const float*)d_in[9];
  const float* o_w       = (const float*)d_in[10];
  const float* o_b       = (const float*)d_in[11];
  float* out = (float*)d_out;

  char* ws = (char*)d_ws;
  const size_t XB = (size_t)NN * CZ * sizeof(_Float16);      // 37,748,736 B
  const size_t NBB = (size_t)NHEAD * NN * sizeof(float);     //  2,359,296 B
  _Float16* Xh = (_Float16*)ws;
  _Float16* WA = (_Float16*)(ws + XB);
  float* NB = (float*)(ws + 2 * XB);
  _Float16* wpack = (_Float16*)(ws + 2 * XB + NBB);  // 5 * 16384 f16
  _Float16* wq = wpack + 0 * 16384;
  _Float16* wk = wpack + 1 * 16384;
  _Float16* wv = wpack + 2 * 16384;
  _Float16* wg = wpack + 3 * 16384;
  _Float16* wo = wpack + 4 * 16384;

  ln_kernel<<<NN / 8, 256, 0, stream>>>(pair_act, ln_g, ln_b, Xh);
  wprep_kernel<<<(5 * 16384) / 256, 256, 0, stream>>>(q_w, k_w, v_w, g_w, o_w, wpack);
  bias_kernel<<<NN / 256, 256, 0, stream>>>(Xh, w2d, NB);
  const size_t shmem =
      (size_t)(NRES * XSTR + NRES * KSTR + CZ * VSTR + 8 * SCRW) * sizeof(_Float16);  // 327,680 B
  attn_kernel<<<NRES, 256, shmem, stream>>>(Xh, pair_mask, NB, wq, wk, wv, wg, g_b, WA);
  out_kernel<<<NN / 128, 256, 0, stream>>>(WA, wo, o_b, out);
}
